// AttDecoder_3874060501757
// MI455X (gfx1250) — compile-verified, matrix operand order checked by reading, not run on hardware
//
#include <hip/hip_runtime.h>
#include <math.h>

typedef __attribute__((ext_vector_type(16))) _Float16 v16h;
typedef __attribute__((ext_vector_type(8)))  _Float16 v8h;
typedef __attribute__((ext_vector_type(8)))  float    v8f;

#define B_   32
#define C_   684
#define H_   16
#define W_   64
#define HW_  1024
#define T_   48
#define E_   256
#define HID_ 256
#define A_   512
#define V_   111
#define LOC_ 432
#define KPAD_CONV 128      // 121 conv taps padded to 4 k-tiles of 32
#define KT_CONV   4
#define CPAD      704      // 684 padded to 22 k-tiles of 32
#define KT_ENC    22
#define NTILES    32       // 512 / 16
#define M_ROWS    (B_*HW_) // 32768

union V16H { v16h v; _Float16 f[16]; };
union V8F  { v8f  v; float    f[8];  };
union V8H  { v8h  v; _Float16 f[8];  };

// Map (kk within 32-k-tile, xx within 16 rows/cols) -> (lane, element) of an
// A or B fragment for v_wmma_f32_16x16x32_f16 (ISA 7.12.2, 16-bit layouts).
__device__ __forceinline__ void frag_pos(int kk, int xx, int& lane, int& elem) {
    int half, e;
    if (kk < 16) { half = kk >> 3; e = kk & 7; }
    else         { int k2 = kk - 16; half = k2 >> 3; e = 8 + (k2 & 7); }
    lane = xx + half * 16;
    elem = e;
}

// ---------------- one-time precompute kernels ----------------

// M[t,a] = sum_c att_conv_k[c,0,t] * att_W[c,a]; store swizzled f16 fragments.
__global__ void k_build_M(const float* __restrict__ attK, const float* __restrict__ attW,
                          _Float16* __restrict__ Msw) {
    int t = blockIdx.x;                              // 0..127 (>=121 are zero pad)
    int a = blockIdx.y * 256 + threadIdx.x;          // 0..511
    float acc = 0.f;
    if (t < 121) {
        for (int c = 0; c < 512; ++c) acc += attK[c * 121 + t] * attW[c * 512 + a];
    }
    int ktile = t >> 5, kk = t & 31, ntile = a >> 4, nn = a & 15, lane, elem;
    frag_pos(kk, nn, lane, elem);
    Msw[(((size_t)ntile * KT_CONV + ktile) * 32 + lane) * 16 + elem] = (_Float16)acc;
}

// Bsw[k,n] = enc_conv_k[n,k] (k padded to 704), swizzled f16 fragments.
__global__ void k_build_Bsw(const float* __restrict__ encK, _Float16* __restrict__ Bsw) {
    int k = blockIdx.x;                              // 0..703
    int a = blockIdx.y * 256 + threadIdx.x;          // 0..511
    float val = (k < C_) ? encK[a * C_ + k] : 0.f;
    int ktile = k >> 5, kk = k & 31, ntile = a >> 4, nn = a & 15, lane, elem;
    frag_pos(kk, nn, lane, elem);
    Bsw[(((size_t)ntile * KT_ENC + ktile) * 32 + lane) * 16 + elem] = (_Float16)val;
}

// masked average over H,W per (b,c); also zero alpha_sum for this call.
__global__ void k_avg(const float* __restrict__ cnn, const float* __restrict__ imask,
                      float* __restrict__ avg, float* __restrict__ alpha_sum) {
    int b = blockIdx.x, tid = threadIdx.x;
    __shared__ float mk[HW_];
    __shared__ float red[256];
    for (int k = tid; k < HW_; k += 256) {
        int h = k >> 6, w = k & 63;
        mk[k] = imask[(size_t)b * 262144 + (h * 16) * 1024 + w * 16];
        alpha_sum[b * HW_ + k] = 0.f;
    }
    __syncthreads();
    float ms = 0.f;
    for (int k = tid; k < HW_; k += 256) ms += mk[k];
    red[tid] = ms; __syncthreads();
    for (int s = 128; s > 0; s >>= 1) { if (tid < s) red[tid] += red[tid + s]; __syncthreads(); }
    float msum = red[0];
    for (int c = tid; c < C_; c += 256) {
        const float* p = cnn + ((size_t)b * C_ + c) * HW_;
        float s = 0.f;
        for (int k = 0; k < HW_; ++k) s += p[k] * mk[k];
        avg[b * C_ + c] = s / msum;
    }
}

// hidden0 = tanh(avg @ W_init + b_init); loc_weighted = loc_pred @ loc_W + loc_b
__global__ void k_hidden0(const float* __restrict__ avg, const float* __restrict__ Winit,
                          const float* __restrict__ binit, const float* __restrict__ locp,
                          const float* __restrict__ locW, const float* __restrict__ locb,
                          float* __restrict__ hidden, float* __restrict__ locwt) {
    int b = blockIdx.x, i = threadIdx.x;
    __shared__ float av[C_];
    __shared__ float lp[LOC_];
    for (int c = i; c < C_;  c += 256) av[c] = avg[b * C_ + c];
    for (int c = i; c < LOC_; c += 256) lp[c] = locp[b * LOC_ + c];
    __syncthreads();
    float s = binit[i];
    for (int c = 0; c < C_; ++c) s += av[c] * Winit[c * HID_ + i];
    hidden[b * HID_ + i] = tanhf(s);
    float l = locb[i];
    for (int c = 0; c < LOC_; ++c) l += lp[c] * locW[c * HID_ + i];
    locwt[b * HID_ + i] = l;
}

// cnn_trans_p: (32768 x 684) @ (684 x 512) WMMA GEMM.
// Output stored in D-fragment order: P[mtile][ntile][lane][8] f16, so the
// per-step consumer streams it with one b128 per lane per n-tile.
__global__ void __launch_bounds__(256) k_cnn_trans(const float* __restrict__ cnn,
                                                   const _Float16* __restrict__ Bsw,
                                                   const float* __restrict__ encB,
                                                   _Float16* __restrict__ P) {
    __shared__ __align__(16) _Float16 Afrag[32 * 16];   // A tile in fragment order
    int tid = threadIdx.x;
    int lane = tid & 31, wave = tid >> 5;
    int mt = blockIdx.x;                 // 0..2047
    int m0 = mt * 16;
    int b = m0 >> 10, hw0 = m0 & 1023;
    int half = lane >> 4, mrow = lane & 15, nlane = lane & 15;
    V8F c[4];
    for (int nt = 0; nt < 4; ++nt)
        for (int e = 0; e < 8; ++e) c[nt].f[e] = 0.f;
    // writer role: this thread fills fragment elements (2*wave, 2*wave+1) of
    // its lane; element pair maps to two consecutive k's of row mrow.
    int elem0 = wave * 2;
    int kloc = (elem0 < 8) ? (half * 8 + elem0) : (16 + half * 8 + (elem0 - 8));
    for (int kt = 0; kt < KT_ENC; ++kt) {
        int k0 = kt * 32;
        __syncthreads();
        {
            int cc = k0 + kloc;
            float v0 = (cc     < C_) ? cnn[((size_t)b * C_ + cc    ) * HW_ + hw0 + mrow] : 0.f;
            float v1 = (cc + 1 < C_) ? cnn[((size_t)b * C_ + cc + 1) * HW_ + hw0 + mrow] : 0.f;
            union { _Float16 h[2]; unsigned u; } pk;
            pk.h[0] = (_Float16)v0;
            pk.h[1] = (_Float16)v1;
            *(unsigned*)&Afrag[lane * 16 + elem0] = pk.u;
        }
        __syncthreads();
        V16H af;
        *(v8h*)&af.f[0] = *(const v8h*)&Afrag[lane * 16];
        *(v8h*)&af.f[8] = *(const v8h*)&Afrag[lane * 16 + 8];
        for (int nt = 0; nt < 4; ++nt) {
            int ntile = wave * 4 + nt;
            v16h bf = *(const v16h*)(Bsw + (((size_t)ntile * KT_ENC + kt) * 32 + lane) * 16);
            c[nt].v = __builtin_amdgcn_wmma_f32_16x16x32_f16(
                false, af.v, false, bf, (short)0, c[nt].v, false, false);
        }
    }
    for (int nt = 0; nt < 4; ++nt) {
        int ntile = wave * 4 + nt;
        int n = ntile * 16 + nlane;
        float bias = encB[n];
        V8H o;
        for (int r = 0; r < 8; ++r) o.f[r] = (_Float16)(c[nt].f[r] + bias);
        *(v8h*)(P + (((size_t)mt * NTILES + ntile) * 32 + lane) * 8) = o.v;
    }
}

// ---------------- per-step kernels ----------------

// GRU cell + query = hidden_new @ hw_W + hw_b (one block per batch row).
__global__ void k_gru(const int* __restrict__ labels, const float* __restrict__ embT,
                      const float* __restrict__ Wih, const float* __restrict__ Whh,
                      const float* __restrict__ bih, const float* __restrict__ bhh,
                      const float* __restrict__ hwW, const float* __restrict__ hwb,
                      float* __restrict__ hidden, float* __restrict__ query,
                      float* __restrict__ embbuf, int t) {
    int b = blockIdx.x, i = threadIdx.x;
    __shared__ float xe[E_], hh[HID_], hn[HID_];
    int tok = (t == 0) ? 1 : labels[b * T_ + (t - 1)];
    xe[i] = embT[tok * E_ + i];
    hh[i] = hidden[b * HID_ + i];
    __syncthreads();
    float gir = bih[i], giz = bih[HID_ + i], gin = bih[2 * HID_ + i];
    const float* wr = Wih + (size_t)i * E_;
    const float* wz = Wih + (size_t)(HID_ + i) * E_;
    const float* wn = Wih + (size_t)(2 * HID_ + i) * E_;
    for (int k = 0; k < E_; ++k) { float x = xe[k]; gir += x * wr[k]; giz += x * wz[k]; gin += x * wn[k]; }
    float ghr = bhh[i], ghz = bhh[HID_ + i], ghn = bhh[2 * HID_ + i];
    const float* vr = Whh + (size_t)i * HID_;
    const float* vz = Whh + (size_t)(HID_ + i) * HID_;
    const float* vn = Whh + (size_t)(2 * HID_ + i) * HID_;
    for (int k = 0; k < HID_; ++k) { float x = hh[k]; ghr += x * vr[k]; ghz += x * vz[k]; ghn += x * vn[k]; }
    float r = 1.f / (1.f + expf(-(gir + ghr)));
    float z = 1.f / (1.f + expf(-(giz + ghz)));
    float n = tanhf(gin + r * ghn);
    float hnew = (1.f - z) * n + z * hh[i];
    hn[i] = hnew;
    __syncthreads();
    hidden[b * HID_ + i] = hnew;
    embbuf[b * E_ + i] = xe[i];
    for (int rep = 0; rep < 2; ++rep) {
        int j = i + rep * 256;
        float q = hwb[j];
        for (int k = 0; k < HID_; ++k) q += hn[k] * hwW[k * A_ + j];
        query[b * A_ + j] = q;
    }
}

// im2col of alpha_sum into (32768 x 128) f16; one b128 store per thread.
__global__ void k_im2col(const float* __restrict__ asum, _Float16* __restrict__ Acol) {
    int t = blockIdx.x * 256 + threadIdx.x;          // 32768*16 threads
    int m = t >> 4, g = t & 15;
    int b = m >> 10, hw = m & 1023;
    int h = hw >> 6, w = hw & 63;
    V8H out;
    for (int q = 0; q < 8; ++q) {
        int kk = g * 8 + q;
        float v = 0.f;
        if (kk < 121) {
            int ii = kk / 11, jj = kk - ii * 11;
            int r = h + ii - 5, cc = w + jj - 5;
            if ((unsigned)r < 16u && (unsigned)cc < 64u) v = asum[b * HW_ + r * 64 + cc];
        }
        out.f[q] = (_Float16)v;
    }
    *(v8h*)(Acol + (size_t)m * KPAD_CONV + g * 8) = out.v;
}

// Fused: conv-as-GEMM (WMMA) + query + cnn_trans_p + tanh + dot(ac_W) -> energy.
__global__ void __launch_bounds__(256) k_att_energy(const _Float16* __restrict__ Acol,
                                                    const _Float16* __restrict__ Msw,
                                                    const _Float16* __restrict__ P,
                                                    const float* __restrict__ query,
                                                    const float* __restrict__ acW,
                                                    const float* __restrict__ acb,
                                                    float* __restrict__ energy) {
    int lane = threadIdx.x & 31;
    int wave = threadIdx.x >> 5;
    int wg = blockIdx.x * 8 + wave;                  // 0..2047 m-tiles
    int m0 = wg * 16;
    int b = m0 >> 10;
    int half = lane >> 4, mrow = lane & 15, nlane = lane & 15;
    int mg = m0 + mrow;
    V16H af[KT_CONV];
    for (int kt = 0; kt < KT_CONV; ++kt) {
        const _Float16* ap = Acol + (size_t)mg * KPAD_CONV + kt * 32 + half * 8;
        *(v8h*)&af[kt].f[0] = *(const v8h*)ap;
        *(v8h*)&af[kt].f[8] = *(const v8h*)(ap + 16);
    }
    float eacc[8];
    for (int r = 0; r < 8; ++r) eacc[r] = 0.f;
    for (int ntile = 0; ntile < NTILES; ++ntile) {
        V8F c;
        for (int e = 0; e < 8; ++e) c.f[e] = 0.f;
        for (int kt = 0; kt < KT_CONV; ++kt) {
            v16h bf = *(const v16h*)(Msw + (((size_t)ntile * KT_CONV + kt) * 32 + lane) * 16);
            c.v = __builtin_amdgcn_wmma_f32_16x16x32_f16(
                false, af[kt].v, false, bf, (short)0, c.v, false, false);
        }
        int n = ntile * 16 + nlane;
        float q  = query[b * A_ + n];
        float aw = acW[n];
        V8H pf;
        pf.v = *(const v8h*)(P + (((size_t)wg * NTILES + ntile) * 32 + lane) * 8);
        for (int r = 0; r < 8; ++r) {
            float val = c.f[r] + q + (float)pf.f[r];
            eacc[r] += tanhf(val) * aw;
        }
    }
    float ab = acb[0];
    for (int r = 0; r < 8; ++r) {
        float v = eacc[r];
        v += __shfl_xor(v, 1, 16);
        v += __shfl_xor(v, 2, 16);
        v += __shfl_xor(v, 4, 16);
        v += __shfl_xor(v, 8, 16);
        if (nlane == 0) energy[m0 + half * 8 + r] = v + ab;
    }
}

// Per-batch masked softmax over H*W; update alpha_sum; write alphas; compact alpha>0.02.
__global__ void k_softmax(const float* __restrict__ energy, const float* __restrict__ imask,
                          float* __restrict__ alpha_sum, float* __restrict__ outA,
                          int* __restrict__ counts, int* __restrict__ cpos,
                          float* __restrict__ cwt, int t) {
    int b = blockIdx.x, tid = threadIdx.x;
    __shared__ float red[256];
    __shared__ int   scan[256];
    float ev[4], mv[4], e[4], a4[4];
    float mx = -1e30f;
    for (int k = 0; k < 4; ++k) {
        int hw = tid + k * 256;
        ev[k] = energy[b * HW_ + hw];
        int h = hw >> 6, w = hw & 63;
        mv[k] = imask[(size_t)b * 262144 + (h * 16) * 1024 + w * 16];
        mx = fmaxf(mx, ev[k]);
    }
    red[tid] = mx; __syncthreads();
    for (int s = 128; s > 0; s >>= 1) { if (tid < s) red[tid] = fmaxf(red[tid], red[tid + s]); __syncthreads(); }
    mx = red[0]; __syncthreads();
    float sum = 0.f;
    for (int k = 0; k < 4; ++k) { e[k] = expf(ev[k] - mx) * mv[k]; sum += e[k]; }
    red[tid] = sum; __syncthreads();
    for (int s = 128; s > 0; s >>= 1) { if (tid < s) red[tid] += red[tid + s]; __syncthreads(); }
    float denom = red[0] + 1e-10f; __syncthreads();
    int cnt = 0;
    for (int k = 0; k < 4; ++k) {
        int hw = tid + k * 256;
        float a = e[k] / denom;
        a4[k] = a;
        alpha_sum[b * HW_ + hw] += a;
        outA[((size_t)b * T_ + t) * HW_ + hw] = a;
        if (a > 0.02f) cnt++;
    }
    scan[tid] = cnt; __syncthreads();
    for (int s = 1; s < 256; s <<= 1) {             // deterministic inclusive scan
        int v = (tid >= s) ? scan[tid - s] : 0;
        __syncthreads();
        scan[tid] += v;
        __syncthreads();
    }
    int off = scan[tid] - cnt;
    for (int k = 0; k < 4; ++k) {
        if (a4[k] > 0.02f) {
            int hw = tid + k * 256;
            cpos[b * HW_ + off] = hw;
            cwt[b * HW_ + off]  = a4[k];
            off++;
        }
    }
    if (tid == 255) counts[b] = scan[255];
}

// ctx[b,c] = sum over compacted positions of w * cnn[b,c,hw]  (sparse reduction).
__global__ void k_ctx(const float* __restrict__ cnn, const int* __restrict__ counts,
                      const int* __restrict__ cpos, const float* __restrict__ cwt,
                      float* __restrict__ ctx) {
    int b = blockIdx.x;
    int c = blockIdx.y * 256 + threadIdx.x;
    __shared__ int   sp[HW_];
    __shared__ float sw[HW_];
    int n = counts[b];
    for (int k = threadIdx.x; k < n; k += 256) { sp[k] = cpos[b * HW_ + k]; sw[k] = cwt[b * HW_ + k]; }
    __syncthreads();
    if (c < C_) {
        const float* p = cnn + ((size_t)b * C_ + c) * HW_;
        float acc = 0.f;
        for (int k = 0; k < n; ++k) acc += sw[k] * p[sp[k]];
        ctx[b * C_ + c] = acc;
    }
}

// out_state = max(h@ws + e@we + ctx@wc + biases, loc_weighted); probs = out_state@out_W + b
__global__ void k_out(const float* __restrict__ hidden, const float* __restrict__ embbuf,
                      const float* __restrict__ ctx, const float* __restrict__ wsW,
                      const float* __restrict__ wsb, const float* __restrict__ weW,
                      const float* __restrict__ web, const float* __restrict__ wcW,
                      const float* __restrict__ wcb, const float* __restrict__ locwt,
                      const float* __restrict__ outW, const float* __restrict__ outb,
                      float* __restrict__ probs, int t) {
    int b = blockIdx.x, i = threadIdx.x;
    __shared__ float hh[HID_], xe[E_], cx[C_], os[HID_];
    hh[i] = hidden[b * HID_ + i];
    xe[i] = embbuf[b * E_ + i];
    for (int k = i; k < C_; k += 256) cx[k] = ctx[b * C_ + k];
    __syncthreads();
    float s = wsb[i] + web[i] + wcb[i];
    for (int k = 0; k < HID_; ++k) s += hh[k] * wsW[k * HID_ + i];
    for (int k = 0; k < E_;  ++k) s += xe[k] * weW[k * HID_ + i];
    for (int k = 0; k < C_;  ++k) s += cx[k] * wcW[k * HID_ + i];
    s = fmaxf(s, locwt[b * HID_ + i]);
    os[i] = s;
    __syncthreads();
    if (i < V_) {
        float p = outb[i];
        for (int k = 0; k < HID_; ++k) p += os[k] * outW[k * V_ + i];
        probs[((size_t)b * T_ + t) * V_ + i] = p;
    }
}

extern "C" void kernel_launch(void* const* d_in, const int* in_sizes, int n_in,
                              void* d_out, int out_size, void* d_ws, size_t ws_size,
                              hipStream_t stream) {
    (void)in_sizes; (void)n_in; (void)out_size; (void)ws_size;
    const float* cnn    = (const float*)d_in[0];
    const int*   labels = (const int*)  d_in[1];
    const float* locp   = (const float*)d_in[2];
    const float* imask  = (const float*)d_in[3];
    const float* Winit  = (const float*)d_in[4];
    const float* binit  = (const float*)d_in[5];
    const float* embT   = (const float*)d_in[6];
    const float* Wih    = (const float*)d_in[7];
    const float* Whh    = (const float*)d_in[8];
    const float* bih    = (const float*)d_in[9];
    const float* bhh    = (const float*)d_in[10];
    const float* hwW    = (const float*)d_in[11];
    const float* hwb    = (const float*)d_in[12];
    const float* attK   = (const float*)d_in[13];
    const float* attW   = (const float*)d_in[14];
    const float* acW    = (const float*)d_in[15];
    const float* acb    = (const float*)d_in[16];
    const float* encK   = (const float*)d_in[17];
    const float* encB   = (const float*)d_in[18];
    const float* wsW    = (const float*)d_in[19];
    const float* wsb    = (const float*)d_in[20];
    const float* weW    = (const float*)d_in[21];
    const float* web    = (const float*)d_in[22];
    const float* wcW    = (const float*)d_in[23];
    const float* wcb    = (const float*)d_in[24];
    const float* locW   = (const float*)d_in[25];
    const float* locb   = (const float*)d_in[26];
    const float* outW   = (const float*)d_in[27];
    const float* outb   = (const float*)d_in[28];

    float* probs  = (float*)d_out;                               // (B,T,V)
    float* alphas = (float*)d_out + (size_t)B_ * T_ * V_;        // (B,T,H,W)

    char* ws = (char*)d_ws;
    size_t off = 0;
    auto alloc = [&](size_t bytes) -> void* {
        void* p = ws + off;
        off = (off + bytes + 255) & ~(size_t)255;
        return p;
    };
    _Float16* P    = (_Float16*)alloc((size_t)M_ROWS * A_ * 2);          // 32 MB, fragment order
    _Float16* Acol = (_Float16*)alloc((size_t)M_ROWS * KPAD_CONV * 2);   // 8 MB
    _Float16* Msw  = (_Float16*)alloc((size_t)NTILES * KT_CONV * 32 * 16 * 2);
    _Float16* Bsw  = (_Float16*)alloc((size_t)NTILES * KT_ENC  * 32 * 16 * 2);
    float* alpha_sum = (float*)alloc((size_t)B_ * HW_ * 4);
    float* hidden    = (float*)alloc((size_t)B_ * HID_ * 4);
    float* query     = (float*)alloc((size_t)B_ * A_ * 4);
    float* energy    = (float*)alloc((size_t)B_ * HW_ * 4);
    float* ctx       = (float*)alloc((size_t)B_ * C_ * 4);
    float* embbuf    = (float*)alloc((size_t)B_ * E_ * 4);
    float* locwt     = (float*)alloc((size_t)B_ * HID_ * 4);
    float* avg       = (float*)alloc((size_t)B_ * C_ * 4);
    int*   counts    = (int*)  alloc((size_t)B_ * 4);
    int*   cpos      = (int*)  alloc((size_t)B_ * HW_ * 4);
    float* cwt       = (float*)alloc((size_t)B_ * HW_ * 4);

    dim3 blk(256);
    k_build_M  <<<dim3(KPAD_CONV, 2), blk, 0, stream>>>(attK, attW, Msw);
    k_build_Bsw<<<dim3(CPAD, 2),      blk, 0, stream>>>(encK, Bsw);
    k_avg      <<<dim3(B_),           blk, 0, stream>>>(cnn, imask, avg, alpha_sum);
    k_hidden0  <<<dim3(B_),           blk, 0, stream>>>(avg, Winit, binit, locp, locW, locb,
                                                        hidden, locwt);
    k_cnn_trans<<<dim3(M_ROWS / 16),  blk, 0, stream>>>(cnn, Bsw, encB, P);

    for (int t = 0; t < T_; ++t) {
        k_gru<<<dim3(B_), blk, 0, stream>>>(labels, embT, Wih, Whh, bih, bhh, hwW, hwb,
                                            hidden, query, embbuf, t);
        k_im2col<<<dim3((M_ROWS * 16) / 256), blk, 0, stream>>>(alpha_sum, Acol);
        k_att_energy<<<dim3(M_ROWS / 16 / 8), blk, 0, stream>>>(Acol, Msw, P, query,
                                                                acW, acb, energy);
        k_softmax<<<dim3(B_), blk, 0, stream>>>(energy, imask, alpha_sum, alphas,
                                                counts, cpos, cwt, t);
        k_ctx<<<dim3(B_, 3), blk, 0, stream>>>(cnn, counts, cpos, cwt, ctx);
        k_out<<<dim3(B_), blk, 0, stream>>>(hidden, embbuf, ctx, wsW, wsb, weW, web,
                                            wcW, wcb, locwt, outW, outb, probs, t);
    }
}
